// Unet_40k_72284299591948
// MI455X (gfx1250) — compile-verified, matrix operand order checked by reading, not run on hardware
//
#include <hip/hip_runtime.h>

// ---------------------------------------------------------------------------
// Spherical U-Net forward for MI455X (gfx1250), all GEMMs on the WMMA pipe.
// v4: software-pipelined GEMM loops (double-buffered fragments) + explicit
// addrspace(1) loads so the inner loop uses global_load_b128 (LOADcnt only)
// instead of flat_load_b128 (LOADcnt+DScnt).
//
// Input flattening (depth-first, setup_inputs() insertion order):
//   [0] x
//   [1..40]   down1..down5, each {W1,b1,g1,be1,W2,b2,g2,be2}
//   [41..80]  up1..up4,     each {Wu,bu, W1,b1,g1,be1,W2,b2,g2,be2}
//   [81..82]  outc {W,b}
//   [83..87]  no0..no4
//   [88..95]  top1,down1,top2,down2,top3,down3,top4,down4
// Workspace: ~40 MB used.
// ---------------------------------------------------------------------------

typedef __attribute__((ext_vector_type(16))) __bf16 v16bf;
typedef __attribute__((ext_vector_type(8)))  __bf16 v8bf;
typedef __attribute__((ext_vector_type(8)))  float  v8f;

// global-address-space pointer types: force global_load_* (not flat_load_*)
typedef const v8bf  __attribute__((address_space(1))) * gp8;
typedef const v16bf __attribute__((address_space(1))) * gp16;

static __device__ __forceinline__ __bf16 f2bf(float x) {
  union { float f; unsigned u; } v; v.f = x;
  unsigned r = v.u + 0x7FFFu + ((v.u >> 16) & 1u);   // round-to-nearest-even
  unsigned short h = (unsigned short)(r >> 16);
  __bf16 b;
  __builtin_memcpy(&b, &h, 2);
  return b;
}

// ---- pack f32 weight [K x Co] into per-tile WMMA B-fragment bf16 layout ----
// tile index = nt*nkt + kt ; within tile: lane*16 + e ; element e of lane maps
// to  k = 32*kt + (e>>3)*16 + (lane>=16)*8 + (e&7),  n = 16*nt + (lane&15).
__global__ void pack_w_kernel(const float* __restrict__ W, __bf16* __restrict__ out,
                              int K, int Co, int nkt, int nnt) {
  int idx = blockIdx.x * blockDim.x + threadIdx.x;
  int total = nkt * nnt * 512;
  if (idx >= total) return;
  int e    = idx & 15;
  int lane = (idx >> 4) & 31;
  int tile = idx >> 9;
  int kt = tile % nkt;
  int nt = tile / nkt;
  int n  = nt * 16 + (lane & 15);
  int khalf = lane >> 4;
  int k  = kt * 32 + ((e >> 3) << 4) + khalf * 8 + (e & 7);
  float v = (k < K && n < Co) ? W[(size_t)k * Co + n] : 0.0f;
  out[idx] = f2bf(v);
}

// ---- one-ring gather conv GEMM: Y[N,Co] = gather7(X,no)[N,7*Ci] @ W + b ----
// X is bf16, Ci % 32 == 0 (each 32-wide K block stays inside one neighbor).
template<int NB>
__global__ void __launch_bounds__(256)
gemm_gather_wmma(const __bf16* __restrict__ X, const int* __restrict__ no,
                 const __bf16* __restrict__ Wpk, const float* __restrict__ bias,
                 float* __restrict__ Y, int N, int Ci, int Co, int nkt) {
  const int lane  = threadIdx.x & 31;
  const int wave  = threadIdx.x >> 5;
  const int mt    = blockIdx.x * 8 + wave;
  const int nt0   = blockIdx.y * NB;
  const int l15   = lane & 15;
  const int khalf = lane >> 4;

  int row = mt * 16 + l15;
  if (row >= N) row = N - 1;
  const int* noRow = no + (size_t)row * 7;

  size_t tb[NB];
#pragma unroll
  for (int t = 0; t < NB; ++t)
    tb[t] = (((size_t)(nt0 + t) * nkt) << 9) + (size_t)lane * 16;

  auto loadA = [&](int kt) -> v16bf {
    int kbase = kt * 32;
    int j     = kbase / Ci;           // uniform within the 32-K block
    int cbase = kbase - j * Ci;
    const __bf16* ap = X + (size_t)noRow[j] * Ci + cbase + khalf * 8;
    v8bf lo = *(gp8)(ap);             // channels +0..7   (16B)
    v8bf hi = *(gp8)(ap + 16);        // channels +16..23 (16B)
    return __builtin_shufflevector(lo, hi, 0,1,2,3,4,5,6,7,8,9,10,11,12,13,14,15);
  };

  v8f acc[NB];
#pragma unroll
  for (int t = 0; t < NB; ++t)
#pragma unroll
    for (int r = 0; r < 8; ++r) acc[t][r] = 0.0f;

  // prologue: stage 0 fragments
  v16bf aC = loadA(0);
  v16bf bC[NB];
#pragma unroll
  for (int t = 0; t < NB; ++t) bC[t] = *(gp16)(Wpk + tb[t]);

  // pipelined main loop: issue kt+1 loads, then MACs on kt's registers
  size_t koff = 0;
  for (int kt = 0; kt < nkt - 1; ++kt) {
    koff += 512;
    v16bf aN = loadA(kt + 1);
    v16bf bN[NB];
#pragma unroll
    for (int t = 0; t < NB; ++t) bN[t] = *(gp16)(Wpk + tb[t] + koff);
#pragma unroll
    for (int t = 0; t < NB; ++t)
      acc[t] = __builtin_amdgcn_wmma_f32_16x16x32_bf16(false, aC, false, bC[t],
                                                       (short)0, acc[t], false, false);
    aC = aN;
#pragma unroll
    for (int t = 0; t < NB; ++t) bC[t] = bN[t];
  }
  // epilogue: last K-block
#pragma unroll
  for (int t = 0; t < NB; ++t)
    acc[t] = __builtin_amdgcn_wmma_f32_16x16x32_bf16(false, aC, false, bC[t],
                                                     (short)0, acc[t], false, false);

#pragma unroll
  for (int t = 0; t < NB; ++t) {
    int n = (nt0 + t) * 16 + l15;
    if (n >= Co) continue;
    float bv = bias[n];
#pragma unroll
    for (int r = 0; r < 8; ++r) {
      int m = mt * 16 + khalf * 8 + r;
      if (m < N) Y[(size_t)m * Co + n] = acc[t][r] + bv;
    }
  }
}

// ---- plain GEMM: Y[N,Co] = Xin[N,K] @ W + b   (Xin bf16, K % 32 == 0) ------
template<int NB>
__global__ void __launch_bounds__(256)
gemm_plain_wmma(const __bf16* __restrict__ Xin, const __bf16* __restrict__ Wpk,
                const float* __restrict__ bias, float* __restrict__ Y,
                int N, int K, int Co, int nkt) {
  const int lane  = threadIdx.x & 31;
  const int wave  = threadIdx.x >> 5;
  const int mt    = blockIdx.x * 8 + wave;
  const int nt0   = blockIdx.y * NB;
  const int l15   = lane & 15;
  const int khalf = lane >> 4;

  int row = mt * 16 + l15;
  if (row >= N) row = N - 1;
  const __bf16* rowp = Xin + (size_t)row * K + khalf * 8;

  size_t tb[NB];
#pragma unroll
  for (int t = 0; t < NB; ++t)
    tb[t] = (((size_t)(nt0 + t) * nkt) << 9) + (size_t)lane * 16;

  auto loadA = [&](int kt) -> v16bf {
    const __bf16* ap = rowp + kt * 32;
    v8bf lo = *(gp8)(ap);
    v8bf hi = *(gp8)(ap + 16);
    return __builtin_shufflevector(lo, hi, 0,1,2,3,4,5,6,7,8,9,10,11,12,13,14,15);
  };

  v8f acc[NB];
#pragma unroll
  for (int t = 0; t < NB; ++t)
#pragma unroll
    for (int r = 0; r < 8; ++r) acc[t][r] = 0.0f;

  v16bf aC = loadA(0);
  v16bf bC[NB];
#pragma unroll
  for (int t = 0; t < NB; ++t) bC[t] = *(gp16)(Wpk + tb[t]);

  size_t koff = 0;
  for (int kt = 0; kt < nkt - 1; ++kt) {
    koff += 512;
    v16bf aN = loadA(kt + 1);
    v16bf bN[NB];
#pragma unroll
    for (int t = 0; t < NB; ++t) bN[t] = *(gp16)(Wpk + tb[t] + koff);
#pragma unroll
    for (int t = 0; t < NB; ++t)
      acc[t] = __builtin_amdgcn_wmma_f32_16x16x32_bf16(false, aC, false, bC[t],
                                                       (short)0, acc[t], false, false);
    aC = aN;
#pragma unroll
    for (int t = 0; t < NB; ++t) bC[t] = bN[t];
  }
#pragma unroll
  for (int t = 0; t < NB; ++t)
    acc[t] = __builtin_amdgcn_wmma_f32_16x16x32_bf16(false, aC, false, bC[t],
                                                     (short)0, acc[t], false, false);

#pragma unroll
  for (int t = 0; t < NB; ++t) {
    int n = (nt0 + t) * 16 + l15;
    if (n >= Co) continue;
    float bv = bias[n];
#pragma unroll
    for (int r = 0; r < 8; ++r) {
      int m = mt * 16 + khalf * 8 + r;
      if (m < N) Y[(size_t)m * Co + n] = acc[t][r] + bv;
    }
  }
}

// ---- first layer: gather x (N,3) -> zero-padded bf16 feat (N,32) -----------
__global__ void gather_pad3_kernel(const float* __restrict__ x, const int* __restrict__ no,
                                   __bf16* __restrict__ feat, int N) {
  int idx = blockIdx.x * blockDim.x + threadIdx.x;
  if (idx >= N * 32) return;
  int c32 = idx & 31;
  int row = idx >> 5;
  float v = 0.0f;
  if (c32 < 21) {
    int j = c32 / 3;
    int c = c32 - j * 3;
    v = x[(size_t)no[(size_t)row * 7 + j] * 3 + c];
  }
  feat[idx] = f2bf(v);
}

// ---- per-column mean/var (biased), deterministic fixed-order f64 reduce ----
__global__ void colstats_kernel(const float* __restrict__ Y, int N, int C,
                                float* __restrict__ mv) {
  int c = blockIdx.x;
  int t = threadIdx.x;
  double s = 0.0, s2 = 0.0;
  for (int r = t; r < N; r += blockDim.x) {
    double v = (double)Y[(size_t)r * C + c];
    s += v; s2 += v * v;
  }
  __shared__ double sh[256], sh2[256];
  sh[t] = s; sh2[t] = s2;
  __syncthreads();
  for (int off = 128; off > 0; off >>= 1) {
    if (t < off) { sh[t] += sh[t + off]; sh2[t] += sh2[t + off]; }
    __syncthreads();
  }
  if (t == 0) {
    double m = sh[0] / (double)N;
    double v = sh2[0] / (double)N - m * m;
    mv[c]     = (float)m;
    mv[C + c] = (float)v;
  }
}

// ---- BatchNorm (batch stats) + LeakyReLU(0.2): f32 in -> bf16 out ----------
__global__ void bn_lrelu_kernel(const float* __restrict__ Y, __bf16* __restrict__ Z,
                                const float* __restrict__ mv,
                                const float* __restrict__ g, const float* __restrict__ be,
                                int N, int C) {
  int idx = blockIdx.x * blockDim.x + threadIdx.x;
  if (idx >= N * C) return;
  int c = idx % C;
  float m = mv[c], v = mv[C + c];
  float y = (Y[idx] - m) * rsqrtf(v + 1e-5f) * g[c] + be[c];
  Z[idx] = f2bf((y > 0.0f) ? y : 0.2f * y);
}

// ---- mean pool over 7-neighborhood of coarse vertices (bf16 -> bf16) -------
__global__ void pool7_kernel(const __bf16* __restrict__ X, const int* __restrict__ no,
                             __bf16* __restrict__ out, int Nc, int C) {
  int idx = blockIdx.x * blockDim.x + threadIdx.x;
  if (idx >= Nc * C) return;
  int c = idx % C;
  int v = idx / C;
  float s = 0.0f;
#pragma unroll
  for (int j = 0; j < 7; ++j)
    s += (float)X[(size_t)no[(size_t)v * 7 + j] * C + c];
  out[idx] = f2bf(s * (1.0f / 7.0f));
}

// ---- upconv rearrange: u = concat(y[top], mean-pair(y[down])), f32->bf16 ---
__global__ void up_rearrange_kernel(const float* __restrict__ Yraw,
                                    const int* __restrict__ top, const int* __restrict__ dwn,
                                    __bf16* __restrict__ u, int raw, int nnew, int co) {
  int idx = blockIdx.x * blockDim.x + threadIdx.x;
  if (idx >= nnew * co) return;
  int c = idx % co;
  int i = idx / co;
  float v;
  if (i < raw) {
    v = Yraw[(size_t)top[i] * co + c];
  } else {
    int p = i - raw;
    v = 0.5f * (Yraw[(size_t)dwn[2 * p] * co + c] + Yraw[(size_t)dwn[2 * p + 1] * co + c]);
  }
  u[idx] = f2bf(v);
}

// ---- channel concat (bf16) -------------------------------------------------
__global__ void concat2_kernel(const __bf16* __restrict__ A, const __bf16* __restrict__ B,
                               __bf16* __restrict__ out, int N, int Ca, int Cb) {
  int C = Ca + Cb;
  int idx = blockIdx.x * blockDim.x + threadIdx.x;
  if (idx >= N * C) return;
  int c = idx % C;
  int r = idx / C;
  out[idx] = (c < Ca) ? A[(size_t)r * Ca + c] : B[(size_t)r * Cb + (c - Ca)];
}

// ===========================================================================
static inline int cdiv(int a, int b) { return (a + b - 1) / b; }

extern "C" void kernel_launch(void* const* d_in, const int* in_sizes, int n_in,
                              void* d_out, int out_size, void* d_ws, size_t ws_size,
                              hipStream_t stream) {
  (void)in_sizes; (void)n_in; (void)out_size; (void)ws_size;

  auto F = [&](int i) { return (const float*)d_in[i]; };
  auto I = [&](int i) { return (const int*)d_in[i]; };

  const float* x = F(0);
  struct Dbl { const float *W1,*b1,*g1,*be1,*W2,*b2,*g2,*be2; };
  auto getDbl = [&](int base) {
    Dbl d;
    d.W1=F(base); d.b1=F(base+1); d.g1=F(base+2); d.be1=F(base+3);
    d.W2=F(base+4); d.b2=F(base+5); d.g2=F(base+6); d.be2=F(base+7);
    return d;
  };
  Dbl dn[5]; for (int i = 0; i < 5; ++i) dn[i] = getDbl(1 + 8 * i);
  const float* upWu[4]; const float* upbu[4]; Dbl upD[4];
  for (int i = 0; i < 4; ++i) {
    int base = 41 + 10 * i;
    upWu[i] = F(base); upbu[i] = F(base + 1);
    upD[i] = getDbl(base + 2);
  }
  const float* ocW = F(81); const float* ocb = F(82);
  const int* no[5]; for (int i = 0; i < 5; ++i) no[i] = I(83 + i);
  const int* topI[4]; const int* dwnI[4];
  for (int i = 0; i < 4; ++i) { topI[i] = I(88 + 2 * i); dwnI[i] = I(89 + 2 * i); }

  // ---- workspace bump allocator ----
  char* wsb = (char*)d_ws;
  size_t cur = 0;
  auto alloc = [&](size_t bytes) -> void* {
    size_t o = (cur + 255) & ~(size_t)255;
    cur = o + bytes;
    return (void*)(wsb + o);
  };

  // ---- pack all weights to bf16 WMMA fragment layout ----
  auto pack = [&](const float* W, int K, int Co) -> __bf16* {
    int nkt = cdiv(K, 32), nnt = cdiv(Co, 16);
    size_t el = (size_t)nkt * nnt * 512;
    __bf16* p = (__bf16*)alloc(el * 2);
    pack_w_kernel<<<cdiv((int)el, 256), 256, 0, stream>>>(W, p, K, Co, nkt, nnt);
    return p;
  };
  const int dCi[5] = {3, 32, 64, 128, 256};
  const int dCo[5] = {32, 64, 128, 256, 512};
  __bf16 *pd1[5], *pd2[5];
  for (int i = 0; i < 5; ++i) {
    pd1[i] = pack(dn[i].W1, 7 * dCi[i], dCo[i]);
    pd2[i] = pack(dn[i].W2, 7 * dCo[i], dCo[i]);
  }
  const int uCi[4] = {512, 256, 128, 64};   // upconv in-ch = concat dbl in-ch
  const int uCo[4] = {256, 128, 64, 32};
  __bf16 *puu[4], *pu1[4], *pu2[4];
  for (int i = 0; i < 4; ++i) {
    puu[i] = pack(upWu[i], uCi[i], 7 * uCo[i]);
    pu1[i] = pack(upD[i].W1, 7 * uCi[i], uCo[i]);
    pu2[i] = pack(upD[i].W2, 7 * uCo[i], uCo[i]);
  }
  __bf16* poc = pack(ocW, 32, 36);

  // ---- activation / scratch buffers ----
  __bf16* x2b = (__bf16*)alloc((size_t)40962 * 32 * 2);
  __bf16* x3b = (__bf16*)alloc((size_t)10242 * 64 * 2);
  __bf16* x4b = (__bf16*)alloc((size_t)2562 * 128 * 2);
  __bf16* x5b = (__bf16*)alloc((size_t)642 * 256 * 2);
  __bf16* x6b = (__bf16*)alloc((size_t)162 * 512 * 2);
  size_t slab = (size_t)40962 * 64;
  float*  F1 = (float*)alloc(slab * 4);    // f32 conv outputs (pre-BN)
  __bf16* B1 = (__bf16*)alloc(slab * 2);   // bf16 GEMM inputs
  __bf16* B2 = (__bf16*)alloc(slab * 2);
  float* stats = (float*)alloc(1024 * 4);

  // ---- launch helpers ----
  auto gemmGather = [&](const __bf16* X, const int* noP, const __bf16* pk, const float* b,
                        float* Y, int N, int Ci, int Co) {
    int nkt = (7 * Ci) / 32;
    int nnt = cdiv(Co, 16);
    dim3 grd(cdiv(cdiv(N, 16), 8), 1, 1);
    if (nnt % 4 == 0) { grd.y = nnt / 4; gemm_gather_wmma<4><<<grd, 256, 0, stream>>>(X, noP, pk, b, Y, N, Ci, Co, nkt); }
    else if (nnt % 2 == 0) { grd.y = nnt / 2; gemm_gather_wmma<2><<<grd, 256, 0, stream>>>(X, noP, pk, b, Y, N, Ci, Co, nkt); }
    else { grd.y = nnt; gemm_gather_wmma<1><<<grd, 256, 0, stream>>>(X, noP, pk, b, Y, N, Ci, Co, nkt); }
  };
  auto gemmPlain = [&](const __bf16* X, const __bf16* pk, const float* b,
                       float* Y, int N, int K, int Co) {
    int nkt = K / 32;
    int nnt = cdiv(Co, 16);
    dim3 grd(cdiv(cdiv(N, 16), 8), 1, 1);
    if (nnt % 4 == 0) { grd.y = nnt / 4; gemm_plain_wmma<4><<<grd, 256, 0, stream>>>(X, pk, b, Y, N, K, Co, nkt); }
    else if (nnt % 2 == 0) { grd.y = nnt / 2; gemm_plain_wmma<2><<<grd, 256, 0, stream>>>(X, pk, b, Y, N, K, Co, nkt); }
    else { grd.y = nnt; gemm_plain_wmma<1><<<grd, 256, 0, stream>>>(X, pk, b, Y, N, K, Co, nkt); }
  };
  auto bnl = [&](const float* Y, __bf16* Z, int N, int C, const float* g, const float* be) {
    colstats_kernel<<<C, 256, 0, stream>>>(Y, N, C, stats);
    bn_lrelu_kernel<<<cdiv(N * C, 256), 256, 0, stream>>>(Y, Z, stats, g, be, N, C);
  };
  auto pool = [&](const __bf16* X, const int* noP, __bf16* out, int Nc, int C) {
    pool7_kernel<<<cdiv(Nc * C, 256), 256, 0, stream>>>(X, noP, out, Nc, C);
  };
  auto rearr = [&](const float* Yraw, int k, __bf16* u, int raw, int nnew, int co) {
    up_rearrange_kernel<<<cdiv(nnew * co, 256), 256, 0, stream>>>(Yraw, topI[k], dwnI[k], u, raw, nnew, co);
  };
  auto cat = [&](const __bf16* A, const __bf16* B, __bf16* out, int N, int Ca, int Cb) {
    concat2_kernel<<<cdiv(N * (Ca + Cb), 256), 256, 0, stream>>>(A, B, out, N, Ca, Cb);
  };

  const int NS[5] = {40962, 10242, 2562, 642, 162};
  __bf16* skip[5] = {x2b, x3b, x4b, x5b, x6b};

  // -------------------- encoder --------------------
  gather_pad3_kernel<<<cdiv(40962 * 32, 256), 256, 0, stream>>>(x, no[0], B1, 40962);
  gemmPlain(B1, pd1[0], dn[0].b1, F1, 40962, 32, 32);
  bnl(F1, B2, 40962, 32, dn[0].g1, dn[0].be1);
  gemmGather(B2, no[0], pd2[0], dn[0].b2, F1, 40962, 32, 32);
  bnl(F1, x2b, 40962, 32, dn[0].g2, dn[0].be2);

  for (int lvl = 1; lvl < 5; ++lvl) {
    pool(skip[lvl - 1], no[lvl - 1], B1, NS[lvl], dCo[lvl - 1]);
    gemmGather(B1, no[lvl], pd1[lvl], dn[lvl].b1, F1, NS[lvl], dCi[lvl], dCo[lvl]);
    bnl(F1, B2, NS[lvl], dCo[lvl], dn[lvl].g1, dn[lvl].be1);
    gemmGather(B2, no[lvl], pd2[lvl], dn[lvl].b2, F1, NS[lvl], dCo[lvl], dCo[lvl]);
    bnl(F1, skip[lvl], NS[lvl], dCo[lvl], dn[lvl].g2, dn[lvl].be2);
  }

  // -------------------- decoder --------------------
  const int raws[4] = {162, 642, 2562, 10242};
  const int news[4] = {642, 2562, 10242, 40962};
  __bf16* h = x6b;                        // (162, 512) bf16
  for (int k = 0; k < 4; ++k) {
    int raw = raws[k], nn = news[k];
    int ci = uCi[k], co = uCo[k];
    int lvl = 3 - k;                      // skip level for this stage
    gemmPlain(h, puu[k], upbu[k], F1, raw, ci, 7 * co);     // (raw, 7co) f32
    rearr(F1, k, B1, raw, nn, co);                          // (nn, co)  bf16
    cat(B1, skip[lvl], B2, nn, co, co);                     // (nn, 2co) bf16
    gemmGather(B2, no[lvl], pu1[k], upD[k].b1, F1, nn, 2 * co, co);
    bnl(F1, B1, nn, co, upD[k].g1, upD[k].be1);
    gemmGather(B1, no[lvl], pu2[k], upD[k].b2, F1, nn, co, co);
    bnl(F1, B2, nn, co, upD[k].g2, upD[k].be2);
    h = B2;
  }

  // -------------------- output head --------------------
  gemmPlain(h, poc, ocb, (float*)d_out, 40962, 32, 36);
}
